// MultiHeadAttention_23716809408750
// MI455X (gfx1250) — compile-verified, matrix operand order checked by reading, not run on hardware
//
#include <hip/hip_runtime.h>
#include <hip/hip_bf16.h>

typedef unsigned short u16;
typedef unsigned int u32;
typedef unsigned long long u64;

typedef __attribute__((ext_vector_type(16))) __bf16 v16bf;
typedef __attribute__((ext_vector_type(8))) float v8f;
typedef __attribute__((ext_vector_type(4))) u32 v4u;
typedef __attribute__((ext_vector_type(8))) u32 v8u;

constexpr int D_MODEL = 1024;
constexpr int NUM_HEADS = 16;
constexpr int DEPTH = 64;      // 1024 / 16
constexpr int SEQ = 2048;
constexpr int BATCH = 4;
constexpr int M_ROWS = BATCH * SEQ;   // 8192

constexpr int KCH = 64;        // GEMM K-chunk staged per TDM transfer
constexpr int LDSTR = 72;      // LDS row stride in bf16 (64 data + 8 pad) = 144B

// ---------------- helpers ----------------

__device__ __forceinline__ u16 f2bf(float f) {
  u32 u = __float_as_uint(f);
  u32 r = 0x7fffu + ((u >> 16) & 1u);   // round-to-nearest-even
  return (u16)((u + r) >> 16);
}

union FragU {
  uint4 u4[2];
  v16bf v;
};

// A-operand (16x32 bf16) fragment load; also used for the symmetric B-operand
// (lane = column). Per CDNA5 ISA 7.12.2: lane = half*16 + m;
//   v0..3 hold K = k0 + half*8 + {0..7}; v4..7 hold K = k0 + 16 + half*8 + {0..7}
// => two contiguous 16B loads per lane.
__device__ __forceinline__ v16bf load_frag(const u16* base, int rowStride, int k0) {
  const int lane = threadIdx.x & 31;
  const int m = lane & 15;
  const int half = lane >> 4;
  const u16* p = base + (size_t)m * rowStride + k0 + half * 8;
  FragU f;
  f.u4[0] = *(const uint4*)(p);
  f.u4[1] = *(const uint4*)(p + 16);
  return f.v;
}

__device__ __forceinline__ v16bf ones_frag() {
  const u32 one2 = 0x3f803f80u;   // two bf16 1.0
  FragU f;
  f.u4[0] = make_uint4(one2, one2, one2, one2);
  f.u4[1] = make_uint4(one2, one2, one2, one2);
  return f.v;
}

__device__ __forceinline__ v8f wmma_bf16(v16bf a, v16bf b, v8f c) {
  return __builtin_amdgcn_wmma_f32_16x16x32_bf16(false, a, false, b, (short)0, c,
                                                 false, false);
}

// Issue a TDM 2D tile load: 64 rows x 64 bf16 from a [1024 x 1024] bf16 tensor
// into LDS at lds_off, with HW padding of 4 DWORDs every 32 DWORDs
// => LDS row stride 144B. (D# per CDNA5 ISA ch.8; 2-group/2D form.)
__device__ __forceinline__ void tdm_load_tile64x64(const u16* gsrc, u32 lds_off) {
  const u64 ga = (u64)(uintptr_t)gsrc;
  v4u g0;
  g0[0] = 1u;                                        // count=1, user descriptor
  g0[1] = lds_off;                                   // lds_addr (bytes)
  g0[2] = (u32)ga;                                   // global_addr[31:0]
  g0[3] = ((u32)(ga >> 32) & 0x01ffffffu) | (2u << 30);  // addr[56:32] | type=2
  v8u g1;
  g1[0] = (1u << 16)      // data_size = 2 bytes
        | (1u << 20)      // pad_enable
        | (4u << 22)      // pad_interval: 2<<4 = 32 DWORDs (= 64 bf16 row)
        | (3u << 25);     // pad_amount: 3+1 = 4 DWORDs (= 8 bf16)
  g1[1] = ((u32)D_MODEL & 0xffffu) << 16;            // tensor_dim0 lo
  g1[2] = ((u32)D_MODEL & 0xffffu) << 16;            // dim0 hi=0 | tensor_dim1 lo
  g1[3] = (u32)KCH << 16;                            // dim1 hi=0 | tile_dim0=64
  g1[4] = 64u;                                       // tile_dim1=64, tile_dim2=0
  g1[5] = (u32)D_MODEL;                              // tensor_dim0_stride lo
  g1[6] = 0u;
  g1[7] = 0u;
  asm volatile("tensor_load_to_lds %0, %1" :: "s"(g0), "s"(g1) : "memory");
}

// ---------------- conversion kernels ----------------

__global__ void cvt_bf16_kernel(const float* __restrict__ src, u16* __restrict__ dst,
                                int n) {
  int i = blockIdx.x * blockDim.x + threadIdx.x;
  if (i < n) dst[i] = f2bf(src[i]);
}

// W is [K, N] fp32 row-major; write Wt [N, K] bf16 row-major.
__global__ void transpose_cvt_kernel(const float* __restrict__ W, u16* __restrict__ Wt,
                                     int K, int N) {
  int i = blockIdx.x * blockDim.x + threadIdx.x;
  if (i < K * N) {
    int k = i / N;
    int nn = i - k * N;
    Wt[(size_t)nn * K + k] = f2bf(W[i]);
  }
}

// ---------------- bf16 WMMA GEMM (TDM double-buffered) ----------------
// C = (A[M_ROWS,K] * Bt[D_MODEL,K]^T + bias) * scale
// Block: 256 threads = 8 waves; block tile 256(M) x 64(N); wave tile 32 x 64.
// out_mode 0: fp32 row-major [M_ROWS, D_MODEL]
// out_mode 1: bf16 head-major   [B, H, S, 64]
// out_mode 2: bf16 head-major-T [B, H, 64, S]
__global__ __launch_bounds__(256) void gemm_bf16_kernel(
    const u16* __restrict__ A, const u16* __restrict__ Bt,
    const float* __restrict__ bias, void* __restrict__ outp, int out_mode,
    float scale) {
  __shared__ u16 sB[2][64 * LDSTR];   // double-buffered B tile, 9216B each
  constexpr int K = D_MODEL;
  constexpr int NCH = K / KCH;        // 16 chunks
  const int wave = threadIdx.x >> 5;
  const int lane = threadIdx.x & 31;
  const int n = lane & 15;
  const int half = lane >> 4;
  const int rowBase = blockIdx.x * 256 + wave * 32;
  const int colBase = blockIdx.y * 64;

  v8f acc[2][4] = {};
  const u16* Arow0 = A + (size_t)rowBase * K;
  const u16* Arow1 = Arow0 + (size_t)16 * K;
  const u16* Btile = Bt + (size_t)colBase * K;
  const u32 lds0 = (u32)(uintptr_t)&sB[0][0];
  const u32 lds1 = (u32)(uintptr_t)&sB[1][0];

  if (wave == 0) tdm_load_tile64x64(Btile, lds0);

  for (int i = 0; i < NCH; ++i) {
    const int buf = i & 1;
    if (wave == 0) {
      if (i + 1 < NCH) {
        tdm_load_tile64x64(Btile + (i + 1) * KCH, buf ? lds0 : lds1);
        __builtin_amdgcn_s_wait_tensorcnt(1);   // chunk i complete; i+1 in flight
      } else {
        __builtin_amdgcn_s_wait_tensorcnt(0);
      }
    }
    __syncthreads();                            // LDS tile visible to all waves

    if (i + 1 < NCH)                            // stream A ahead into L2
      __builtin_prefetch(Arow0 + (size_t)lane * K + (i + 1) * KCH, 0, 3);

    const u16* sBb = &sB[buf][0];
#pragma unroll
    for (int kk = 0; kk < 2; ++kk) {
      const int kg = i * KCH + kk * 32;
      v16bf a0 = load_frag(Arow0, K, kg);
      v16bf a1 = load_frag(Arow1, K, kg);
#pragma unroll
      for (int t = 0; t < 4; ++t) {
        v16bf b = load_frag(sBb + t * 16 * LDSTR, LDSTR, kk * 32);
        acc[0][t] = wmma_bf16(a0, b, acc[0][t]);
        acc[1][t] = wmma_bf16(a1, b, acc[1][t]);
      }
    }
    __syncthreads();                            // done reading buf before reuse
  }

#pragma unroll
  for (int g = 0; g < 2; ++g) {
#pragma unroll
    for (int t = 0; t < 4; ++t) {
      const int col = colBase + t * 16 + n;
      const float bv = bias[col];
      const int h = col >> 6;
      const int dd = col & 63;
#pragma unroll
      for (int r = 0; r < 8; ++r) {
        const int row = rowBase + g * 16 + r + 8 * half;
        const float val = (acc[g][t][r] + bv) * scale;
        if (out_mode == 0) {
          ((float*)outp)[(size_t)row * D_MODEL + col] = val;
        } else {
          const int b = row / SEQ;
          const int s = row - b * SEQ;
          size_t off;
          if (out_mode == 1)
            off = (((size_t)b * NUM_HEADS + h) * SEQ + s) * DEPTH + dd;
          else
            off = (((size_t)b * NUM_HEADS + h) * DEPTH + dd) * SEQ + s;
          ((u16*)outp)[off] = f2bf(val);
        }
      }
    }
  }
}

// ---------------- flash attention ----------------
// Qh (pre-scaled by 1/sqrt(depth)), Kh: bf16 [B, H, S, 64]; Vt: bf16 [B, H, 64, S]
// mask fp32 [B, S]; ctxb out: bf16 [B*S, D_MODEL] for the final GEMM.
// Softmax denominator accumulated as an extra WMMA (P x ones) so it rescales
// with the context accumulator; no cross-lane sum reduction needed.
__global__ __launch_bounds__(256) void attn_kernel(
    const u16* __restrict__ Qh, const u16* __restrict__ Kh,
    const u16* __restrict__ Vt, const float* __restrict__ mask,
    u16* __restrict__ ctxb) {
  __shared__ u16 sP[8][16 * 40];   // per-wave P tile (16 rows x 32 cols, padded)
  const int wave = threadIdx.x >> 5;
  const int lane = threadIdx.x & 31;
  const int n = lane & 15;
  const int half = lane >> 4;

  constexpr int QTILES = SEQ / 128;            // 16
  const int bh = blockIdx.x / QTILES;          // 0..B*H-1
  const int qblk = blockIdx.x - bh * QTILES;
  const int b = bh / NUM_HEADS;
  const int h = bh - b * NUM_HEADS;
  const int q0 = qblk * 128 + wave * 16;

  const u16* Qbase = Qh + ((size_t)bh * SEQ + q0) * DEPTH;
  const u16* Kbase = Kh + (size_t)bh * SEQ * DEPTH;
  const u16* Vbase = Vt + (size_t)bh * DEPTH * SEQ;
  const float* mrow = mask + (size_t)b * SEQ;

  const v16bf aq0 = load_frag(Qbase, DEPTH, 0);
  const v16bf aq1 = load_frag(Qbase, DEPTH, 32);
  const v16bf vone = ones_frag();

  v8f ctx[4] = {};
  v8f ctxS = {};                 // softmax denominator per row (replicated cols)
  float rm[8];
#pragma unroll
  for (int r = 0; r < 8; ++r) rm[r] = -3.0e38f;

  u16* myP = &sP[wave][0];

  for (int kv = 0; kv < SEQ; kv += 32) {
    // ---- logits tile 16 x 32 = Q . K^T (Q pre-scaled) ----
    v8f c0 = {}, c1 = {};
    {
      v16bf bk00 = load_frag(Kbase + (size_t)kv * DEPTH, DEPTH, 0);
      v16bf bk01 = load_frag(Kbase + (size_t)kv * DEPTH, DEPTH, 32);
      v16bf bk10 = load_frag(Kbase + (size_t)(kv + 16) * DEPTH, DEPTH, 0);
      v16bf bk11 = load_frag(Kbase + (size_t)(kv + 16) * DEPTH, DEPTH, 32);
      c0 = wmma_bf16(aq0, bk00, c0);
      c0 = wmma_bf16(aq1, bk01, c0);
      c1 = wmma_bf16(aq0, bk10, c1);
      c1 = wmma_bf16(aq1, bk11, c1);
    }

    const float m0 = mrow[kv + n] * -1e9f;
    const float m1 = mrow[kv + 16 + n] * -1e9f;

    // ---- online softmax: max-reduce per row, exp, rescale accumulators ----
#pragma unroll
    for (int r = 0; r < 8; ++r) {
      float l0 = c0[r] + m0;
      float l1 = c1[r] + m1;
      float t = fmaxf(l0, l1);
#pragma unroll
      for (int off = 1; off < 16; off <<= 1)
        t = fmaxf(t, __shfl_xor(t, off, 16));
      const float mnew = fmaxf(rm[r], t);
      const float corr = __expf(rm[r] - mnew);
      const float p0 = __expf(l0 - mnew);
      const float p1 = __expf(l1 - mnew);
      rm[r] = mnew;
#pragma unroll
      for (int t4 = 0; t4 < 4; ++t4) ctx[t4][r] = ctx[t4][r] * corr;
      ctxS[r] = ctxS[r] * corr;
      const int row = r + 8 * half;
      myP[row * 40 + n] = f2bf(p0);          // C-layout -> LDS
      myP[row * 40 + 16 + n] = f2bf(p1);
    }

    // ---- P (16x32) back as A fragment; denominator + PV WMMAs ----
    const v16bf ap = load_frag(myP, 40, 0);  // wave-private; DS in-order per wave
    ctxS = wmma_bf16(ap, vone, ctxS);        // rowsum(P) accumulated per row
#pragma unroll
    for (int t4 = 0; t4 < 4; ++t4) {
      v16bf bv = load_frag(Vbase + (size_t)(t4 * 16) * SEQ, SEQ, kv);
      ctx[t4] = wmma_bf16(ap, bv, ctx[t4]);
    }
  }

  // ---- normalize and store context ----
#pragma unroll
  for (int t4 = 0; t4 < 4; ++t4) {
#pragma unroll
    for (int r = 0; r < 8; ++r) {
      const int row = q0 + r + 8 * half;
      const float val = ctx[t4][r] / ctxS[r];
      const int col = h * DEPTH + t4 * 16 + n;
      ctxb[((size_t)b * SEQ + row) * D_MODEL + col] = f2bf(val);
    }
  }
}

// ---------------- launch ----------------

extern "C" void kernel_launch(void* const* d_in, const int* in_sizes, int n_in,
                              void* d_out, int out_size, void* d_ws, size_t ws_size,
                              hipStream_t stream) {
  const float* q    = (const float*)d_in[0];
  const float* kin  = (const float*)d_in[1];
  const float* vin  = (const float*)d_in[2];
  const float* mask = (const float*)d_in[3];
  const float* Wq   = (const float*)d_in[4];
  const float* bq   = (const float*)d_in[5];
  const float* Wk   = (const float*)d_in[6];
  const float* bk   = (const float*)d_in[7];
  const float* Wv   = (const float*)d_in[8];
  const float* bv   = (const float*)d_in[9];
  const float* Wo   = (const float*)d_in[10];
  const float* bo   = (const float*)d_in[11];

  constexpr size_t MB = 1ull << 20;
  char* ws = (char*)d_ws;
  u16* qb   = (u16*)(ws + 0 * MB);     // 16 MiB each: [8192,1024] bf16
  u16* kb   = (u16*)(ws + 16 * MB);
  u16* vb   = (u16*)(ws + 32 * MB);
  u16* wqT  = (u16*)(ws + 48 * MB);    // 2 MiB each: [1024,1024] bf16
  u16* wkT  = (u16*)(ws + 50 * MB);
  u16* wvT  = (u16*)(ws + 52 * MB);
  u16* woT  = (u16*)(ws + 54 * MB);
  u16* Qh   = (u16*)(ws + 56 * MB);    // [B,H,S,64] bf16 (pre-scaled)
  u16* Kh   = (u16*)(ws + 72 * MB);    // [B,H,S,64] bf16
  u16* Vt   = (u16*)(ws + 88 * MB);    // [B,H,64,S] bf16
  u16* ctxb = (u16*)(ws + 104 * MB);   // [8192,1024] bf16

  const int nX = M_ROWS * D_MODEL;           // 8388608
  const int nW = D_MODEL * D_MODEL;          // 1048576

  cvt_bf16_kernel<<<nX / 256, 256, 0, stream>>>(q, qb, nX);
  cvt_bf16_kernel<<<nX / 256, 256, 0, stream>>>(kin, kb, nX);
  cvt_bf16_kernel<<<nX / 256, 256, 0, stream>>>(vin, vb, nX);

  transpose_cvt_kernel<<<nW / 256, 256, 0, stream>>>(Wq, wqT, D_MODEL, D_MODEL);
  transpose_cvt_kernel<<<nW / 256, 256, 0, stream>>>(Wk, wkT, D_MODEL, D_MODEL);
  transpose_cvt_kernel<<<nW / 256, 256, 0, stream>>>(Wv, wvT, D_MODEL, D_MODEL);
  transpose_cvt_kernel<<<nW / 256, 256, 0, stream>>>(Wo, woT, D_MODEL, D_MODEL);

  dim3 gg(M_ROWS / 256, D_MODEL / 64);
  // 1/sqrt(DEPTH) folded into the Q projection epilogue (exact in bf16).
  gemm_bf16_kernel<<<gg, 256, 0, stream>>>(qb, wqT, bq, Qh, 1, 0.125f);
  gemm_bf16_kernel<<<gg, 256, 0, stream>>>(kb, wkT, bk, Kh, 1, 1.0f);
  gemm_bf16_kernel<<<gg, 256, 0, stream>>>(vb, wvT, bv, Vt, 2, 1.0f);

  attn_kernel<<<BATCH * NUM_HEADS * (SEQ / 128), 256, 0, stream>>>(Qh, Kh, Vt, mask,
                                                                   ctxb);

  gemm_bf16_kernel<<<gg, 256, 0, stream>>>(ctxb, woT, bo, d_out, 0, 1.0f);
}
